// Gemma4Attention_75376676045209
// MI455X (gfx1250) — matrix operand order repaired and checked by reading, not compiled, not run on hardware
//
#include <hip/hip_runtime.h>
#include <hip/hip_bf16.h>

// ---------------------------------------------------------------------------
// Gemma-style attention layer for MI455X (gfx1250, wave32, WMMA).
// All matmuls run through one batched bf16 WMMA GEMM (f32 accumulate):
//   out[m,n] = sum_k A[m,k] * W[n,k]     (A, W both K-contiguous bf16)
// Wave tile 32x64 (2 A-frags x 4 B-frags = 8 WMMA / K-step) to raise
// FLOP-per-L2-byte; block = 8 waves -> 64x256 tile.
// ---------------------------------------------------------------------------

#define DEV static __device__ __forceinline__

typedef __attribute__((ext_vector_type(8)))  float  v8f;
typedef __attribute__((ext_vector_type(8)))  __bf16 v8bf;
typedef __attribute__((ext_vector_type(16))) __bf16 v16bf;

// Model constants
static constexpr int   T_   = 1024;
static constexpr int   HID_ = 2048;
static constexpr int   NH_  = 16;
static constexpr int   NKV_ = 4;
static constexpr int   HD_  = 512;
static constexpr int   ROT_ = 128;
static constexpr float EPS_ = 1e-6f;
static constexpr float LOG_THETA = 13.815510558f; // ln(1e6)

// fp32 -> bf16 round-to-nearest-even
DEV unsigned short f2bf(float f) {
    unsigned int u = __float_as_uint(f);
    u += 0x7FFFu + ((u >> 16) & 1u);
    return (unsigned short)(u >> 16);
}

DEV v16bf load_a_frag(const unsigned short* p) {
    // 16-bit A 16x32, k order chosen here (B uses the same; not the ISA table order): per lane, contiguous K chunk [k0..k0+15]
    v8bf lo = *(const v8bf*)(p);
    v8bf hi = *(const v8bf*)(p + 8);
    return __builtin_shufflevector(lo, hi, 0,1,2,3,4,5,6,7,8,9,10,11,12,13,14,15);
}

DEV v8f wmma_bf16(v16bf a, v16bf b, v8f c) {
    return __builtin_amdgcn_wmma_f32_16x16x32_bf16(
        /*neg_a=*/false, a, /*neg_b=*/false, b,
        /*c_mod=*/(short)0, c, /*reuse_a=*/false, /*reuse_b=*/false);
}

// ---------------------------------------------------------------------------
// Batched GEMM: C[b] = A[b] * W[b/wdiv]^T   (bf16 in, f32 out)
// Block = 256 threads = 8 waves arranged 2(M) x 4(N); block tile 64 x 256.
// Per-wave tile 32x64: acc[2][4] v8f, 8 x v_wmma_f32_16x16x32_bf16 per K-step.
// All M,N,K used here divide the tiles exactly (no bounds divergence; EXEC
// stays all-ones inside WMMA waves as the ISA requires).
// ---------------------------------------------------------------------------
__global__ __launch_bounds__(256)
void gemm_bf16_kernel(float* __restrict__ C,
                      const unsigned short* __restrict__ A,
                      const unsigned short* __restrict__ W,
                      int M, int N, int K, int lda, int ldw, int ldc,
                      long long strideA, long long strideW, long long strideC,
                      int wdiv)
{
    const int lane = threadIdx.x & 31;
    const int wave = threadIdx.x >> 5;
    const int lh   = lane >> 4;   // half-wave (0/1)
    const int ln   = lane & 15;

    const int b = blockIdx.z;
    const unsigned short* Ab = A + (long long)b * strideA;
    const unsigned short* Wb = W + (long long)(b / wdiv) * strideW;
    float* Cb = C + (long long)b * strideC;

    const int m0 = blockIdx.y * 64  + (wave >> 2) * 32;  // wave tile: 32 rows
    const int n0 = blockIdx.x * 256 + (wave & 3) * 64;   // wave tile: 64 cols
    if (m0 >= M || n0 >= N) return;

    v8f acc[2][4];
    #pragma unroll
    for (int i = 0; i < 2; ++i)
        #pragma unroll
        for (int j = 0; j < 4; ++j) acc[i][j] = {};

    const unsigned short* aRow[2];
    const unsigned short* wRow[4];
    #pragma unroll
    for (int i = 0; i < 2; ++i)
        aRow[i] = Ab + (long long)(m0 + 16 * i + ln) * lda + lh * 16;
    #pragma unroll
    for (int j = 0; j < 4; ++j)
        wRow[j] = Wb + (long long)(n0 + 16 * j + ln) * ldw + lh * 16;

    // Warm the L2/WGP$ path for the first few lines of each stream
    // (emits gfx1250 global_prefetch_b8; speculative, OOB-safe).
    #pragma unroll
    for (int i = 0; i < 2; ++i) __builtin_prefetch(aRow[i], 0, 3);
    #pragma unroll
    for (int j = 0; j < 4; ++j) __builtin_prefetch(wRow[j], 0, 3);

    #pragma unroll 2
    for (int kb = 0; kb < K; kb += 32) {
        v16bf af[2], bf[4];
        #pragma unroll
        for (int i = 0; i < 2; ++i) af[i] = load_a_frag(aRow[i] + kb);
        #pragma unroll
        for (int j = 0; j < 4; ++j) bf[j] = *(const v16bf*)(wRow[j] + kb);
        #pragma unroll
        for (int i = 0; i < 2; ++i)
            #pragma unroll
            for (int j = 0; j < 4; ++j)
                acc[i][j] = wmma_bf16(af[i], bf[j], acc[i][j]);
    }

    // C/D layout: lane l, reg r -> (m = r + 8*(l>>4), n = l&15)
    #pragma unroll
    for (int i = 0; i < 2; ++i) {
        #pragma unroll
        for (int j = 0; j < 4; ++j) {
            #pragma unroll
            for (int r = 0; r < 8; ++r) {
                const int mr = m0 + 16 * i + r + 8 * lh;
                Cb[(long long)mr * ldc + n0 + 16 * j + ln] = acc[i][j][r];
            }
        }
    }
}

// ---------------------------------------------------------------------------
// f32 -> bf16 converter, 4 elements/thread (all sizes divisible by 1024)
// ---------------------------------------------------------------------------
__global__ __launch_bounds__(256)
void cvt4_kernel(unsigned short* __restrict__ dst,
                 const float* __restrict__ src, long long n4)
{
    long long i = (long long)blockIdx.x * blockDim.x + threadIdx.x;
    if (i >= n4) return;
    float4 v = ((const float4*)src)[i];
    ushort4 o;
    o.x = f2bf(v.x); o.y = f2bf(v.y); o.z = f2bf(v.z); o.w = f2bf(v.w);
    ((ushort4*)dst)[i] = o;
}

// ---------------------------------------------------------------------------
// Fused RMS-norm + partial RoPE + layout transform.
//   blockIdx.y in [0,NH)        : q head  -> qb[h][t][d]   (bf16)
//   blockIdx.y in [NH,NH+NKV)   : kv head -> kb[kv][t][d]  (norm*w + rope)
//                                           vT[kv][d][t]   (norm only, transposed)
// 256 threads, each handles d and d+256 of the 512-wide head.
// ---------------------------------------------------------------------------
__global__ __launch_bounds__(256)
void normrope_kernel(const float* __restrict__ qf, const float* __restrict__ kf,
                     const float* __restrict__ qnw, const float* __restrict__ knw,
                     unsigned short* __restrict__ qb,
                     unsigned short* __restrict__ kb,
                     unsigned short* __restrict__ vT)
{
    const int t    = blockIdx.x;
    const int head = blockIdx.y;
    const int tid  = threadIdx.x;
    const bool isq = head < NH_;

    const float* src = isq ? (qf + (long long)t * (NH_  * HD_) + head          * HD_)
                           : (kf + (long long)t * (NKV_ * HD_) + (head - NH_)  * HD_);

    __shared__ float red[256];
    float x0 = src[tid], x1 = src[tid + 256];
    red[tid] = x0 * x0 + x1 * x1;
    __syncthreads();
    for (int s = 128; s > 0; s >>= 1) {
        if (tid < s) red[tid] += red[tid + s];
        __syncthreads();
    }
    const float scale = rsqrtf(red[0] / (float)HD_ + EPS_);

    const float* nw = isq ? qnw : knw;
    #pragma unroll
    for (int pass = 0; pass < 2; ++pass) {
        const int d = tid + pass * 256;
        const float y  = src[d] * scale;           // plain RMS-norm value
        float yw = y * nw[d];                      // with learned scale
        if (d < ROT_) {                            // partial rotary on first 128 dims
            const int   i    = d & 63;
            const float invf = __expf(-((float)(2 * i) / (float)ROT_) * LOG_THETA);
            const float ang  = (float)t * invf;
            const float c = cosf(ang), s = sinf(ang);
            const float pw = src[d ^ 64] * scale * nw[d ^ 64];
            yw = (d < 64) ? (yw * c - pw * s) : (yw * c + pw * s);
        }
        if (isq) {
            qb[(long long)head * T_ * HD_ + (long long)t * HD_ + d] = f2bf(yw);
        } else {
            const int kv = head - NH_;
            kb[(long long)kv * T_ * HD_ + (long long)t * HD_ + d] = f2bf(yw);
            vT[(long long)kv * HD_ * T_ + (long long)d * T_ + t]  = f2bf(y); // v: no scale, no rope
        }
    }
}

// ---------------------------------------------------------------------------
// Causal row softmax over S[h][t][0..T): keep s<=t, zero the masked tail so
// the PV GEMM can run dense over K=T. Writes P as bf16.
// ---------------------------------------------------------------------------
__global__ __launch_bounds__(256)
void softmax_kernel(const float* __restrict__ S, unsigned short* __restrict__ P)
{
    const int t = blockIdx.x, h = blockIdx.y, tid = threadIdx.x;
    const float* row = S + ((long long)h * T_ + t) * (long long)T_;
    unsigned short* prow = P + ((long long)h * T_ + t) * (long long)T_;
    __shared__ float red[256];

    float m = -3.0e38f;
    for (int s = tid; s <= t; s += 256) m = fmaxf(m, row[s]);
    red[tid] = m; __syncthreads();
    for (int st = 128; st > 0; st >>= 1) {
        if (tid < st) red[tid] = fmaxf(red[tid], red[tid + st]);
        __syncthreads();
    }
    const float mx = red[0];
    __syncthreads();

    float sum = 0.f;
    for (int s = tid; s <= t; s += 256) sum += __expf(row[s] - mx);
    red[tid] = sum; __syncthreads();
    for (int st = 128; st > 0; st >>= 1) {
        if (tid < st) red[tid] += red[tid + st];
        __syncthreads();
    }
    const float inv = 1.0f / red[0];

    for (int s = tid; s < T_; s += 256) {
        const float p = (s <= t) ? __expf(row[s] - mx) * inv : 0.0f;
        prow[s] = f2bf(p);
    }
}

// ---------------------------------------------------------------------------
extern "C" void kernel_launch(void* const* d_in, const int* in_sizes, int n_in,
                              void* d_out, int out_size, void* d_ws, size_t ws_size,
                              hipStream_t stream)
{
    (void)in_sizes; (void)n_in; (void)out_size; (void)ws_size;
    const float* x   = (const float*)d_in[0];  // (T, HID)
    const float* qw  = (const float*)d_in[1];  // (NH*HD, HID)
    const float* kw  = (const float*)d_in[2];  // (NKV*HD, HID)
    const float* ow  = (const float*)d_in[3];  // (HID, NH*HD)
    const float* qnw = (const float*)d_in[4];  // (HD)
    const float* knw = (const float*)d_in[5];  // (HD)
    // d_in[6] = input_pos == arange(T): positions are implicit (t = row index)
    float* out = (float*)d_out;                // (T, HID) f32

    // ---- workspace layout (bytes, 256-aligned by construction) ----
    char* ws = (char*)d_ws;
    size_t off = 0;
    auto alloc = [&](size_t bytes) { size_t o = off; off += (bytes + 255) & ~(size_t)255; return o; };
    const size_t XB  = alloc((size_t)T_ * HID_ * 2);            // x bf16
    const size_t QWB = alloc((size_t)NH_ * HD_ * HID_ * 2);     // q_w bf16
    const size_t KWB = alloc((size_t)NKV_ * HD_ * HID_ * 2);    // k_w bf16
    const size_t OWB = alloc((size_t)HID_ * NH_ * HD_ * 2);     // o_w bf16
    const size_t QF  = alloc((size_t)T_ * NH_ * HD_ * 4);       // q proj f32 (reused as attn f32)
    const size_t KF  = alloc((size_t)T_ * NKV_ * HD_ * 4);      // k proj f32
    const size_t QB  = alloc((size_t)NH_ * T_ * HD_ * 2);       // q normed bf16 [h][t][d]
    const size_t KB  = alloc((size_t)NKV_ * T_ * HD_ * 2);      // k normed bf16 [kv][t][d]
    const size_t VT  = alloc((size_t)NKV_ * HD_ * T_ * 2);      // v normed bf16 [kv][d][t]
    const size_t SB  = alloc((size_t)NH_ * T_ * T_ * 4);        // scores f32
    const size_t PB  = alloc((size_t)NH_ * T_ * T_ * 2);        // probs bf16
    const size_t AF  = QF;                                      // attn out f32 [t][h*HD+d] (aliases dead QF)
    const size_t AB  = alloc((size_t)T_ * NH_ * HD_ * 2);       // attn out bf16

    unsigned short* xb  = (unsigned short*)(ws + XB);
    unsigned short* qwb = (unsigned short*)(ws + QWB);
    unsigned short* kwb = (unsigned short*)(ws + KWB);
    unsigned short* owb = (unsigned short*)(ws + OWB);
    float*          qf  = (float*)(ws + QF);
    float*          kf  = (float*)(ws + KF);
    unsigned short* qb  = (unsigned short*)(ws + QB);
    unsigned short* kb  = (unsigned short*)(ws + KB);
    unsigned short* vT  = (unsigned short*)(ws + VT);
    float*          S   = (float*)(ws + SB);
    unsigned short* P   = (unsigned short*)(ws + PB);
    float*          af  = (float*)(ws + AF);
    unsigned short* ab  = (unsigned short*)(ws + AB);

    const dim3 blk(256);
    auto cvt = [&](unsigned short* dst, const float* src, long long n) {
        long long n4 = n >> 2;
        cvt4_kernel<<<dim3((unsigned)((n4 + 255) / 256)), blk, 0, stream>>>(dst, src, n4);
    };

    // 1) down-convert activations and weights to bf16
    cvt(xb,  x,  (long long)T_ * HID_);
    cvt(qwb, qw, (long long)NH_ * HD_ * HID_);
    cvt(kwb, kw, (long long)NKV_ * HD_ * HID_);
    cvt(owb, ow, (long long)HID_ * NH_ * HD_);

    // 2) q = x @ q_w^T   (1024 x 8192 x 2048)
    gemm_bf16_kernel<<<dim3(8192 / 256, 1024 / 64, 1), blk, 0, stream>>>(
        qf, xb, qwb, T_, NH_ * HD_, HID_, HID_, HID_, NH_ * HD_, 0, 0, 0, 1);

    // 3) k = x @ k_w^T   (1024 x 2048 x 2048)
    gemm_bf16_kernel<<<dim3(2048 / 256, 1024 / 64, 1), blk, 0, stream>>>(
        kf, xb, kwb, T_, NKV_ * HD_, HID_, HID_, HID_, NKV_ * HD_, 0, 0, 0, 1);

    // 4) RMS-norm + RoPE + head-major/transposed layouts
    normrope_kernel<<<dim3(T_, NH_ + NKV_), blk, 0, stream>>>(qf, kf, qnw, knw, qb, kb, vT);

    // 5) scores[h] = q[h] @ k[h/4]^T   (batched over 16 heads, 1024x1024x512)
    gemm_bf16_kernel<<<dim3(1024 / 256, 1024 / 64, NH_), blk, 0, stream>>>(
        S, qb, kb, T_, T_, HD_, HD_, HD_, T_,
        (long long)T_ * HD_, (long long)T_ * HD_, (long long)T_ * T_, NH_ / NKV_);

    // 6) causal softmax -> P (bf16, masked tail zeroed)
    softmax_kernel<<<dim3(T_, NH_), blk, 0, stream>>>(S, P);

    // 7) attn[h] = P[h] @ vT[h/4]^T   (batched, 1024x512x1024) -> af[t][h*HD+d]
    gemm_bf16_kernel<<<dim3(512 / 256, 1024 / 64, NH_), blk, 0, stream>>>(
        af, P, vT, T_, HD_, T_, T_, T_, NH_ * HD_,
        (long long)T_ * T_, (long long)HD_ * T_, (long long)HD_, NH_ / NKV_);

    // 8) convert attention output to bf16
    cvt(ab, af, (long long)T_ * NH_ * HD_);

    // 9) out = attn @ o_w^T   (1024 x 2048 x 8192) -> d_out (f32)
    gemm_bf16_kernel<<<dim3(2048 / 256, 1024 / 64, 1), blk, 0, stream>>>(
        out, ab, owb, T_, HID_, NH_ * HD_, NH_ * HD_, NH_ * HD_, HID_, 0, 0, 0, 1);
}
